// JunNet4_55825984913474
// MI455X (gfx1250) — matrix operand order repaired and checked by reading, not compiled, not run on hardware
//
#include <hip/hip_runtime.h>

// ---------------------------------------------------------------------------
// Types for CDNA5 WMMA
// ---------------------------------------------------------------------------
typedef __bf16 bf16_t;
typedef bf16_t v16bf __attribute__((ext_vector_type(16)));
typedef float  v8f   __attribute__((ext_vector_type(8)));

#define TBM 64
#define TBN 64
#define TBK 64
#define PLD 66   // padded LDS row stride in bf16 elems (33 dwords, coprime w/ 64 banks)

union FragBF { v16bf v; unsigned int u[8]; };

static __device__ __forceinline__ float apply_act(float v, int act) {
  if (act == 1) return fmaxf(v, 0.f);
  if (act == 2) return 0.5f * v * (1.f + erff(v * 0.70710678118654752f));
  return v;
}

// ---------------------------------------------------------------------------
// Generic WMMA GEMM:  C[z][M][N] = act( scale[n]*(A[z] x Bop[z]) + shift[n] + R[z] )
//   A: [M][K] f32 (row stride lda), batch offset = z1*saz1 + z2*saz2
//   B: b_nk ? [N][K] : [K][N]  (row stride ldb)
//   grid.z = Z1*Z2, z1 = z/Z2, z2 = z%Z2
// ---------------------------------------------------------------------------
__global__ __launch_bounds__(128) void wmma_gemm(
    const float* __restrict__ A, long lda, long saz1, long saz2,
    const float* __restrict__ B, long ldb, int b_nk, long sbz1, long sbz2,
    float* C, long ldc, long scz1, long scz2,
    const float* R, long srz1, long srz2,
    const float* __restrict__ scale, const float* __restrict__ shift,
    int M, int N, int K, int Z2, int act)
{
  __shared__ __align__(16) bf16_t sA[TBM * PLD];
  __shared__ __align__(16) bf16_t sB[TBN * PLD];
  const int t = threadIdx.x;
  const int lane = t & 31, w = t >> 5;
  const int half = lane >> 4, lr = lane & 15;
  const int wm = (w >> 1) * 32, wn = (w & 1) * 32;
  const int m0 = blockIdx.y * TBM, n0 = blockIdx.x * TBN;
  const int z1 = blockIdx.z / Z2, z2 = blockIdx.z % Z2;
  A += (long)z1 * saz1 + (long)z2 * saz2;
  B += (long)z1 * sbz1 + (long)z2 * sbz2;
  C += (long)z1 * scz1 + (long)z2 * scz2;
  if (R) R += (long)z1 * srz1 + (long)z2 * srz2;

  const int lcol = t & 63;   // scalar path: fixed K-column this thread stages
  const int lrow0 = t >> 6;  // scalar path: rows r = 2*i + lrow0
  const int qrow = t >> 4;          // vector path: base row (16 quads per row)
  const int qcol = (t & 15) * 4;    // vector path: K-column of this thread's quad

  const bool alignedA = ((lda & 3) == 0);
  const bool alignedB = ((ldb & 3) == 0);

  v8f acc[2][2] = {};

  for (int k0 = 0; k0 < K; k0 += TBK) {
    const bool fullK = (k0 + TBK) <= K;
    // ---- stage A tile ----
    if (fullK && alignedA && (m0 + TBM) <= M) {
      const float* Ab = A + (long)m0 * lda + k0;
#pragma unroll
      for (int i = 0; i < 8; ++i) {
        int r = qrow + i * 8;
        const float4 vv = *(const float4*)&Ab[(long)r * lda + qcol];
        bf16_t* d = &sA[r * PLD + qcol];
        d[0] = (bf16_t)vv.x; d[1] = (bf16_t)vv.y;
        d[2] = (bf16_t)vv.z; d[3] = (bf16_t)vv.w;
      }
      if (k0 + TBK < K)  // pull next K-tile toward the WGP while we compute
        __builtin_prefetch(&Ab[(long)lcol * lda + TBK], 0, 1);
    } else {
      const int gk = k0 + lcol;
      const bool okk = (gk < K);
#pragma unroll
      for (int i = 0; i < 32; ++i) {
        int r = 2 * i + lrow0;
        int gm = m0 + r;
        bool ok = okk && (gm < M);
        long idx = ok ? ((long)gm * lda + gk) : 0;
        float v = A[idx];
        sA[r * PLD + lcol] = (bf16_t)(ok ? v : 0.f);
      }
    }
    // ---- stage B tile as [N][K] ----
    if (fullK && b_nk && alignedB && (n0 + TBN) <= N) {
      const float* Bb = B + (long)n0 * ldb + k0;
#pragma unroll
      for (int i = 0; i < 8; ++i) {
        int r = qrow + i * 8;
        const float4 vv = *(const float4*)&Bb[(long)r * ldb + qcol];
        bf16_t* d = &sB[r * PLD + qcol];
        d[0] = (bf16_t)vv.x; d[1] = (bf16_t)vv.y;
        d[2] = (bf16_t)vv.z; d[3] = (bf16_t)vv.w;
      }
      if (k0 + TBK < K)
        __builtin_prefetch(&Bb[(long)lcol * ldb + TBK], 0, 1);
    } else {
      const int gk = k0 + lcol;
      const bool okk = (gk < K);
#pragma unroll
      for (int i = 0; i < 32; ++i) {
        int r = 2 * i + lrow0;
        int gn = n0 + r;
        bool ok = okk && (gn < N);
        long idx = ok ? (b_nk ? ((long)gn * ldb + gk) : ((long)gk * ldb + gn)) : 0;
        float v = B[idx];
        sB[r * PLD + lcol] = (bf16_t)(ok ? v : 0.f);
      }
    }
    __syncthreads();
#pragma unroll
    for (int kk = 0; kk < 2; ++kk) {
      FragBF fa[2], fb[2];
#pragma unroll
      for (int s = 0; s < 2; ++s) {
        const int ar = wm + s * 16 + lr;
        const int br = wn + s * 16 + lr;
#pragma unroll
        for (int j = 0; j < 8; ++j) {
          int ka = kk * 32 + 2 * j + 8 * half + ((j >= 4) ? 8 : 0); // A 16x32 layout
          int kb = kk * 32 + 2 * j + 8 * half + ((j >= 4) ? 8 : 0); // B 32x16 layout
          fa[s].u[j] = *(const unsigned int*)&sA[ar * PLD + ka];
          fb[s].u[j] = *(const unsigned int*)&sB[br * PLD + kb];
        }
      }
#pragma unroll
      for (int i = 0; i < 2; ++i)
#pragma unroll
        for (int j = 0; j < 2; ++j)
          acc[i][j] = __builtin_amdgcn_wmma_f32_16x16x32_bf16(
              false, fa[i].v, false, fb[j].v, (short)0, acc[i][j], false, false);
    }
    __syncthreads();
  }

#pragma unroll
  for (int i = 0; i < 2; ++i) {
#pragma unroll
    for (int j = 0; j < 2; ++j) {
      int n = n0 + wn + j * 16 + lr;
      if (n >= N) continue;
      float sc = scale ? scale[n] : 1.f;
      float sh = shift ? shift[n] : 0.f;
#pragma unroll
      for (int r = 0; r < 8; ++r) {
        int m = m0 + wm + i * 16 + r + 8 * half;
        if (m >= M) continue;
        float v = acc[i][j][r] * sc + sh;
        if (R) v += R[(long)m * ldc + n];
        C[(long)m * ldc + n] = apply_act(v, act);
      }
    }
  }
}

// ---------------------------------------------------------------------------
// 3x3 conv as implicit GEMM over NHWC activations.
//   m = (b,y,x);  k = ci*9 + (ky*3+kx);  weights: torch [Cout][Cin][3][3] == [N][K]
//   H, W are powers of two: wsh = log2(W), hwsh = log2(H*W)
// ---------------------------------------------------------------------------
__global__ __launch_bounds__(128) void wmma_conv3(
    const float* __restrict__ In, int H, int W, int wsh, int hwsh, int Cin,
    const float* __restrict__ Wt, float* C, const float* R,
    const float* __restrict__ scale, const float* __restrict__ shift,
    int M, int N, int act)
{
  const int K = Cin * 9;
  __shared__ __align__(16) bf16_t sA[TBM * PLD];
  __shared__ __align__(16) bf16_t sB[TBN * PLD];
  const int t = threadIdx.x;
  const int lane = t & 31, w = t >> 5;
  const int half = lane >> 4, lr = lane & 15;
  const int wm = (w >> 1) * 32, wn = (w & 1) * 32;
  const int m0 = blockIdx.y * TBM, n0 = blockIdx.x * TBN;
  const int lcol = t & 63;
  const int lrow0 = t >> 6;
  const int qrow = t >> 4;
  const int qcol = (t & 15) * 4;
  const bool alignedB = ((K & 3) == 0);

  v8f acc[2][2] = {};

  for (int k0 = 0; k0 < K; k0 += TBK) {
    const int gk = k0 + lcol;
    const bool okk = (gk < K);
    // per-thread K-column decomposition (once per K step)
    const int ci = gk / 9;
    const int tap = gk - ci * 9;
    const int dy = tap / 3 - 1;
    const int dx = tap - (tap / 3) * 3 - 1;
    // ---- stage A tile (inherent 3x3-halo gather; scalar, branchless) ----
#pragma unroll
    for (int i = 0; i < 32; ++i) {
      int r = 2 * i + lrow0;
      int gm = m0 + r;
      int b = gm >> hwsh;
      int rem = gm & ((1 << hwsh) - 1);
      int y = rem >> wsh;
      int x = rem & (W - 1);
      int yy = y + dy, xx = x + dx;
      bool ok = okk && (gm < M) && (yy >= 0) && (yy < H) && (xx >= 0) && (xx < W);
      long pix = ((long)b << hwsh) + ((long)yy << wsh) + xx;
      long idx = ok ? (pix * Cin + ci) : 0;
      float v = In[idx];
      sA[r * PLD + lcol] = (bf16_t)(ok ? v : 0.f);
    }
    // ---- stage B tile [N][K] (weights; vectorized when aligned+interior) ----
    if (((k0 + TBK) <= K) && alignedB && (n0 + TBN) <= N) {
      const float* Bb = Wt + (long)n0 * K + k0;
#pragma unroll
      for (int i = 0; i < 8; ++i) {
        int r = qrow + i * 8;
        const float4 vv = *(const float4*)&Bb[(long)r * K + qcol];
        bf16_t* d = &sB[r * PLD + qcol];
        d[0] = (bf16_t)vv.x; d[1] = (bf16_t)vv.y;
        d[2] = (bf16_t)vv.z; d[3] = (bf16_t)vv.w;
      }
      if (k0 + TBK < K)
        __builtin_prefetch(&Bb[(long)lcol * K + TBK], 0, 1);
    } else {
#pragma unroll
      for (int i = 0; i < 32; ++i) {
        int r = 2 * i + lrow0;
        int gn = n0 + r;
        bool ok = okk && (gn < N);
        long idx = ok ? ((long)gn * K + gk) : 0;
        float v = Wt[idx];
        sB[r * PLD + lcol] = (bf16_t)(ok ? v : 0.f);
      }
    }
    __syncthreads();
#pragma unroll
    for (int kk = 0; kk < 2; ++kk) {
      FragBF fa[2], fb[2];
#pragma unroll
      for (int s = 0; s < 2; ++s) {
        const int ar = wm + s * 16 + lr;
        const int br = wn + s * 16 + lr;
#pragma unroll
        for (int j = 0; j < 8; ++j) {
          int ka = kk * 32 + 2 * j + 8 * half + ((j >= 4) ? 8 : 0);
          int kb = kk * 32 + 2 * j + 8 * half + ((j >= 4) ? 8 : 0);
          fa[s].u[j] = *(const unsigned int*)&sA[ar * PLD + ka];
          fb[s].u[j] = *(const unsigned int*)&sB[br * PLD + kb];
        }
      }
#pragma unroll
      for (int i = 0; i < 2; ++i)
#pragma unroll
        for (int j = 0; j < 2; ++j)
          acc[i][j] = __builtin_amdgcn_wmma_f32_16x16x32_bf16(
              false, fa[i].v, false, fb[j].v, (short)0, acc[i][j], false, false);
    }
    __syncthreads();
  }

#pragma unroll
  for (int i = 0; i < 2; ++i) {
#pragma unroll
    for (int j = 0; j < 2; ++j) {
      int n = n0 + wn + j * 16 + lr;
      if (n >= N) continue;
      float sc = scale ? scale[n] : 1.f;
      float sh = shift ? shift[n] : 0.f;
#pragma unroll
      for (int r = 0; r < 8; ++r) {
        int m = m0 + wm + i * 16 + r + 8 * half;
        if (m >= M) continue;
        float v = acc[i][j][r] * sc + sh;
        if (R) v += R[(long)m * (long)N + n];
        C[(long)m * (long)N + n] = apply_act(v, act);
      }
    }
  }
}

// ---------------------------------------------------------------------------
// Helper kernels
// ---------------------------------------------------------------------------
__global__ void prep_bn_k(const float* g, const float* b, const float* m,
                          const float* v, const float* cb,
                          float* scale, float* shift, int n) {
  int i = blockIdx.x * blockDim.x + threadIdx.x;
  if (i < n) {
    float s = g[i] * rsqrtf(v[i] + 1e-5f);
    scale[i] = s;
    shift[i] = b[i] - m[i] * s + (cb ? cb[i] * s : 0.f);
  }
}

__global__ void nchw2nhwc_k(const float* in, float* out, int B, int C, int H, int W) {
  long total = (long)B * C * H * W;
  long idx = (long)blockIdx.x * blockDim.x + threadIdx.x;
  if (idx >= total) return;
  int c = idx % C; long t = idx / C;
  int x = t % W; t /= W;
  int y = t % H; int b = t / H;
  out[idx] = in[(((long)b * C + c) * H + y) * W + x];
}

__global__ void nhwc2nchw_k(const float* in, float* out, int B, int C, int H, int W) {
  long total = (long)B * C * H * W;
  long idx = (long)blockIdx.x * blockDim.x + threadIdx.x;
  if (idx >= total) return;
  int x = idx % W; long t = idx / W;
  int y = t % H; t /= H;
  int c = t % C; int b = t / C;
  out[idx] = in[(((long)b * H + y) * W + x) * C + c];
}

__global__ void pool12_k(const float* in, float* out, int B, int H, int W, int C) {
  int OW = W >> 1;
  long total = (long)B * H * OW * C;
  long idx = (long)blockIdx.x * blockDim.x + threadIdx.x;
  if (idx >= total) return;
  int c = idx % C; long t = idx / C;
  int x = t % OW; t /= OW;
  int y = t % H; int b = t / H;
  const float* p = in + (((long)b * H + y) * W + 2 * x) * C + c;
  out[idx] = fmaxf(p[0], p[C]);
}

__global__ void upsample_bl_k(const float* in, float* out,
                              int B, int H, int W, int C, int sh, int sw) {
  int OH = H * sh, OW = W * sw;
  long total = (long)B * OH * OW * C;
  long idx = (long)blockIdx.x * blockDim.x + threadIdx.x;
  if (idx >= total) return;
  int c = idx % C; long t = idx / C;
  int ox = t % OW; t /= OW;
  int oy = t % OH; int b = t / OH;
  float fy = (oy + 0.5f) / (float)sh - 0.5f;
  float fx = (ox + 0.5f) / (float)sw - 0.5f;
  int y0 = (int)floorf(fy); float wy = fy - (float)y0;
  int x0 = (int)floorf(fx); float wx = fx - (float)x0;
  int y0c = min(max(y0, 0), H - 1), y1c = min(max(y0 + 1, 0), H - 1);
  int x0c = min(max(x0, 0), W - 1), x1c = min(max(x0 + 1, 0), W - 1);
  const float* base = in + (long)b * H * W * C + c;
  float v00 = base[((long)y0c * W + x0c) * C];
  float v01 = base[((long)y0c * W + x1c) * C];
  float v10 = base[((long)y1c * W + x0c) * C];
  float v11 = base[((long)y1c * W + x1c) * C];
  out[idx] = (1.f - wy) * ((1.f - wx) * v00 + wx * v01)
           +        wy  * ((1.f - wx) * v10 + wx * v11);
}

__global__ void im2patch_k(const float* in, float* out,
                           int B, int H, int W, int C, int P, int GH, int GW) {
  int PP = P * P;
  long total = (long)B * GH * GW * C * PP;
  long idx = (long)blockIdx.x * blockDim.x + threadIdx.x;
  if (idx >= total) return;
  int kk = idx % (C * PP); long t = idx / (C * PP);
  int g = t % (GH * GW); int b = t / (GH * GW);
  int c = kk / PP; int r = kk - c * PP;
  int py = r / P, px = r - py * P;
  int gy = g / GW, gx = g - gy * GW;
  out[idx] = in[(((long)b * H + gy * P + py) * W + gx * P + px) * C + c];
}

__global__ void add_cls_pos_k(float* tok, const float* cls, const float* pos,
                              int B, int n) {
  long total = (long)B * n * 512;
  long idx = (long)blockIdx.x * blockDim.x + threadIdx.x;
  if (idx >= total) return;
  int c = idx % 512; long t = idx / 512;
  int i = t % n;
  if (i == 0) tok[idx] = cls[c] + pos[c];
  else        tok[idx] += pos[(long)i * 512 + c];
}

__global__ __launch_bounds__(256) void layernorm512_k(const float* in, float* out,
                                                      const float* g, const float* b) {
  long row = blockIdx.x;
  const float* xr = in + row * 512;
  float* yr = out + row * 512;
  int t = threadIdx.x;
  float a0 = xr[t], a1 = xr[t + 256];
  __shared__ float red[256];
  red[t] = a0 + a1; __syncthreads();
  for (int k = 128; k > 0; k >>= 1) { if (t < k) red[t] += red[t + k]; __syncthreads(); }
  float mean = red[0] * (1.f / 512.f); __syncthreads();
  float d0 = a0 - mean, d1 = a1 - mean;
  red[t] = d0 * d0 + d1 * d1; __syncthreads();
  for (int k = 128; k > 0; k >>= 1) { if (t < k) red[t] += red[t + k]; __syncthreads(); }
  float inv = rsqrtf(red[0] * (1.f / 512.f) + 1e-5f);
  yr[t]       = d0 * inv * g[t]       + b[t];
  yr[t + 256] = d1 * inv * g[t + 256] + b[t + 256];
}

__global__ __launch_bounds__(256) void softmax_rows_k(float* data, int n, float sc) {
  long row = blockIdx.x;
  float* x = data + row * (long)n;
  int t = threadIdx.x;
  __shared__ float red[256];
  float mx = -1e30f;
  for (int i = t; i < n; i += 256) mx = fmaxf(mx, x[i] * sc);
  red[t] = mx; __syncthreads();
  for (int k = 128; k > 0; k >>= 1) { if (t < k) red[t] = fmaxf(red[t], red[t + k]); __syncthreads(); }
  mx = red[0]; __syncthreads();
  float sum = 0.f;
  for (int i = t; i < n; i += 256) { float e = expf(x[i] * sc - mx); x[i] = e; sum += e; }
  red[t] = sum; __syncthreads();
  for (int k = 128; k > 0; k >>= 1) { if (t < k) red[t] += red[t + k]; __syncthreads(); }
  float inv = 1.f / red[0];
  for (int i = t; i < n; i += 256) x[i] *= inv;
}

__global__ void copy2d_k(float* dst, long ldd, long dz,
                         const float* src, long lds_, long sz, int rows, int cols) {
  long total = (long)rows * cols;
  long i = (long)blockIdx.x * blockDim.x + threadIdx.x;
  if (i >= total) return;
  int r = (int)(i / cols), c = (int)(i % cols);
  int z = blockIdx.z;
  dst[(long)z * dz + (long)r * ldd + c] = src[(long)z * sz + (long)r * lds_ + c];
}

// ---------------------------------------------------------------------------
// Host side
// ---------------------------------------------------------------------------
static inline int idivup(long a, long b) { return (int)((a + b - 1) / b); }
static inline int ilog2(int v) { int s = 0; while ((1 << s) < v) ++s; return s; }

struct BNP { const float *b, *g, *m, *v; };
struct FLp { BNP bn1, bn2, skip_bn; const float *skip_w, *w1, *w2; bool skip; };
struct UBp { BNP bn1, bn2; const float *skip_b, *skip_w, *w1, *w2; bool skip; };
struct DECp { UBp blk; const float *up_b; BNP up_bn; const float *up_w; };
struct LYRp { const float *mlp_b1, *mlp_b2, *mlp_w1, *mlp_w2, *n1_b, *n1_g,
                          *n2_b, *n2_g, *out_b, *out_w, *qkv_w; };
struct VITp { const float *cls, *in_b, *in_w; LYRp L[2]; const float *pos; };

extern "C" void kernel_launch(void* const* d_in, const int* in_sizes, int n_in,
                              void* d_out, int out_size, void* d_ws, size_t ws_size,
                              hipStream_t stream) {
  (void)in_sizes; (void)n_in; (void)out_size; (void)ws_size;
  const int B = 4;

  // ---- parameter extraction (JAX pytree order: dict keys sorted) ----
  int cur = 0;
  auto nx = [&]() { return (const float*)d_in[cur++]; };
  const float* X = nx();  // "x" first (setup_inputs dict order), then "params"

  auto getbn = [&]() { BNP r; r.b = nx(); r.g = nx(); r.m = nx(); r.v = nx(); return r; };
  auto getFL = [&](bool skip) {
    FLp f; f.skip = skip; f.bn1 = getbn(); f.bn2 = getbn();
    if (skip) { f.skip_bn = getbn(); f.skip_w = nx(); }
    f.w1 = nx(); f.w2 = nx(); return f;
  };
  auto getUB = [&](bool skip) {
    UBp u; u.skip = skip; u.bn1 = getbn(); u.bn2 = getbn();
    if (skip) { u.skip_b = nx(); u.skip_w = nx(); }
    u.w1 = nx(); u.w2 = nx(); return u;
  };
  auto getDEC = [&]() {
    DECp d; d.blk = getUB(true); d.up_b = nx(); d.up_bn = getbn(); d.up_w = nx(); return d;
  };
  auto getLYR = [&]() {
    LYRp l; l.mlp_b1 = nx(); l.mlp_b2 = nx(); l.mlp_w1 = nx(); l.mlp_w2 = nx();
    l.n1_b = nx(); l.n1_g = nx(); l.n2_b = nx(); l.n2_g = nx();
    l.out_b = nx(); l.out_w = nx(); l.qkv_w = nx(); return l;
  };
  auto getVIT = [&]() {
    VITp v; v.cls = nx(); v.in_b = nx(); v.in_w = nx();
    v.L[0] = getLYR(); v.L[1] = getLYR(); v.pos = nx(); return v;
  };

  FLp fb1[3]; fb1[0] = getFL(true);  fb1[1] = getFL(true);  fb1[2] = getFL(false);
  FLp fb2[3]; fb2[0] = getFL(false); fb2[1] = getFL(true);  fb2[2] = getFL(false);
  const float* fc_b = nx();
  const float* fc_w = nx();
  DECp fin = getDEC();
  UBp res1 = getUB(true);
  DECp up0 = getDEC(), up1 = getDEC(), up2 = getDEC();
  VITp vit0 = getVIT(), vit1 = getVIT(), vit2 = getVIT();

  // ---- workspace layout (float elems) ----
  float* ws = (float*)d_ws;
  size_t off = 0;
  auto alloc = [&](size_t ne) { float* p = ws + off; off += ne; return p; };
  float* XH    = alloc((size_t)B * 64 * 512 * 5);
  float* T0    = alloc(33554432);
  float* T1    = alloc(33554432);
  float* T2    = alloc(16777216);
  float* T3    = alloc(16777216);
  float* DOWN1 = alloc((size_t)B * 64 * 256 * 32);
  float* DOWN2 = alloc((size_t)B * 64 * 128 * 64);
  float* VS0   = alloc((size_t)B * 128 * 512);
  float* VS1   = alloc((size_t)B * 256 * 512);
  float* VS2   = alloc((size_t)B * 512 * 512);
  float* D0    = alloc((size_t)B * 64 * 64 * 256);
  float* D1    = alloc((size_t)B * 64 * 64 * 256);
  float* D2    = alloc((size_t)B * 64 * 64 * 256);
  float* TOK   = alloc((size_t)B * 513 * 512);
  float* TLN   = alloc((size_t)B * 513 * 512);
  float* QKV   = alloc((size_t)B * 513 * 1536);
  float* SC    = alloc((size_t)16 * 513 * 513);
  float* AT    = alloc((size_t)B * 513 * 512);
  float* H1    = alloc((size_t)B * 513 * 512);
  float* PATCH = alloc(2097152);
  float* BNB   = alloc(80 * 1024);
  float* FCB   = alloc((size_t)B * 64 * 512 * 20);

  // ---- launch helpers ----
  auto gemm = [&](const float* A, long lda, long saz1, long saz2,
                  const float* Bm, long ldb, bool bnk, long sbz1, long sbz2,
                  float* C, long ldc, long scz1, long scz2,
                  const float* R, long srz1, long srz2,
                  const float* scale, const float* shift,
                  int M, int N, int K, int Z1, int Z2, int act) {
    dim3 g(idivup(N, TBN), idivup(M, TBM), Z1 * Z2), blk(128);
    wmma_gemm<<<g, blk, 0, stream>>>(A, lda, saz1, saz2, Bm, ldb, bnk ? 1 : 0,
                                     sbz1, sbz2, C, ldc, scz1, scz2,
                                     R, srz1, srz2, scale, shift,
                                     M, N, K, Z2, act);
  };
  auto conv3 = [&](const float* in, int H, int W, int Cin, const float* w, int Cout,
                   float* out, const float* resid, const float* scale,
                   const float* shift, int act) {
    int M = B * H * W;
    int wsh = ilog2(W), hwsh = wsh + ilog2(H);
    dim3 g(idivup(Cout, TBN), idivup(M, TBM), 1), blk(128);
    wmma_conv3<<<g, blk, 0, stream>>>(in, H, W, wsh, hwsh, Cin, w, out, resid,
                                      scale, shift, M, Cout, act);
  };
  auto conv1x1 = [&](const float* in, int M, int Cin, const float* w, int Cout,
                     float* out, const float* resid, const float* scale,
                     const float* shift, int act) {
    gemm(in, Cin, 0, 0, w, Cin, true, 0, 0, out, Cout, 0, 0,
         resid, 0, 0, scale, shift, M, Cout, Cin, 1, 1, act);
  };
  int bnslot = 0;
  auto prep = [&](const BNP& bn, const float* cb, int n) -> const float* {
    float* s = BNB + (size_t)bnslot * 1024; bnslot++;
    prep_bn_k<<<dim3(idivup(n, 256)), dim3(256), 0, stream>>>(
        bn.g, bn.b, bn.m, bn.v, cb, s, s + 512, n);
    return s;
  };

  auto fl_fwd = [&](const FLp& f, const float* in, int H, int W, int Cin, int Cout,
                    float* out, float* tmp, float* idn) {
    int M = B * H * W;
    const float* resid = in;
    if (f.skip) {
      const float* ss = prep(f.skip_bn, nullptr, Cout);
      conv1x1(in, M, Cin, f.skip_w, Cout, idn, nullptr, ss, ss + 512, 0);
      resid = idn;
    }
    const float* s1 = prep(f.bn1, nullptr, Cout);
    conv3(in, H, W, Cin, f.w1, Cout, tmp, nullptr, s1, s1 + 512, 1);
    const float* s2 = prep(f.bn2, nullptr, Cout);
    conv3(tmp, H, W, Cout, f.w2, Cout, out, resid, s2, s2 + 512, 1);
  };
  auto ub_fwd = [&](const UBp& u, const float* in, int H, int W, int Cin, int Cout,
                    float* out, float* tmp, float* idn) {
    int M = B * H * W;
    const float* resid = in;
    if (u.skip) {
      conv1x1(in, M, Cin, u.skip_w, Cout, idn, nullptr, nullptr, u.skip_b, 0);
      resid = idn;
    }
    const float* s1 = prep(u.bn1, nullptr, Cout);
    conv3(in, H, W, Cin, u.w1, Cout, tmp, nullptr, s1, s1 + 512, 1);
    const float* s2 = prep(u.bn2, nullptr, Cout);
    conv3(tmp, H, W, Cout, u.w2, Cout, out, resid, s2, s2 + 512, 1);
  };
  auto dec_fwd = [&](const DECp& d, const float* in, int H, int W, int sh, int sw,
                     int Cenc, int Cdown, int Cout,
                     float* bufUp, float* bufConv, float* tmp, float* idn, float* out) {
    int OH = H * sh, OW = W * sw;
    long tot = (long)B * OH * OW * Cenc;
    upsample_bl_k<<<dim3(idivup(tot, 256)), dim3(256), 0, stream>>>(
        in, bufUp, B, H, W, Cenc, sh, sw);
    const float* su = prep(d.up_bn, d.up_b, Cdown);
    conv3(bufUp, OH, OW, Cenc, d.up_w, Cdown, bufConv, nullptr, su, su + 512, 1);
    ub_fwd(d.blk, bufConv, OH, OW, Cdown, Cout, out, tmp, idn);
  };

  auto vit_fwd = [&](const VITp& V, const float* in, int H, int W, int Cin, int P,
                     float* vsOut) {
    int GH = H / P, GW = W / P, np = GH * GW, n = np + 1, idim = Cin * P * P;
    long ptot = (long)B * np * idim;
    im2patch_k<<<dim3(idivup(ptot, 256)), dim3(256), 0, stream>>>(
        in, PATCH, B, H, W, Cin, P, GH, GW);
    gemm(PATCH, idim, (long)np * idim, 0,
         V.in_w, idim, true, 0, 0,
         TOK + 512, 512, (long)n * 512, 0,
         nullptr, 0, 0, nullptr, V.in_b,
         np, 512, idim, B, 1, 0);
    long ttot = (long)B * n * 512;
    add_cls_pos_k<<<dim3(idivup(ttot, 256)), dim3(256), 0, stream>>>(
        TOK, V.cls, V.pos, B, n);
    int Rrows = B * n;
    long nn = (long)n * n;
    for (int l = 0; l < 2; ++l) {
      const LYRp& L = V.L[l];
      layernorm512_k<<<dim3(Rrows), dim3(256), 0, stream>>>(TOK, TLN, L.n1_g, L.n1_b);
      gemm(TLN, 512, 0, 0, L.qkv_w, 512, true, 0, 0, QKV, 1536, 0, 0,
           nullptr, 0, 0, nullptr, nullptr, Rrows, 1536, 512, 1, 1, 0);
      // scores[b][h] = q @ k^T
      gemm(QKV, 1536, (long)n * 1536, 128,
           QKV + 512, 1536, true, (long)n * 1536, 128,
           SC, n, 4 * nn, nn,
           nullptr, 0, 0, nullptr, nullptr, n, n, 128, B, 4, 0);
      softmax_rows_k<<<dim3(B * 4 * n), dim3(256), 0, stream>>>(
          SC, n, 0.08838834764831845f);
      // attn = P @ V  (heads concatenated into AT[...,512])
      gemm(SC, n, 4 * nn, nn,
           QKV + 1024, 1536, false, (long)n * 1536, 128,
           AT, 512, (long)n * 512, 128,
           nullptr, 0, 0, nullptr, nullptr, n, 128, n, B, 4, 0);
      gemm(AT, 512, 0, 0, L.out_w, 512, true, 0, 0, TOK, 512, 0, 0,
           TLN, 0, 0, nullptr, L.out_b, Rrows, 512, 512, 1, 1, 0);
      layernorm512_k<<<dim3(Rrows), dim3(256), 0, stream>>>(TOK, TLN, L.n2_g, L.n2_b);
      gemm(TLN, 512, 0, 0, L.mlp_w1, 512, true, 0, 0, H1, 512, 0, 0,
           nullptr, 0, 0, nullptr, L.mlp_b1, Rrows, 512, 512, 1, 1, 2);
      gemm(H1, 512, 0, 0, L.mlp_w2, 512, true, 0, 0, TOK, 512, 0, 0,
           TLN, 0, 0, nullptr, L.mlp_b2, Rrows, 512, 512, 1, 1, 0);
    }
    long vtot = (long)np * 512;
    copy2d_k<<<dim3(idivup(vtot, 256), 1, B), dim3(256), 0, stream>>>(
        vsOut, 512, (long)np * 512, TOK + 512, 512, (long)n * 512, np, 512);
  };

  // ======================= forward pipeline =======================
  {
    long tot = (long)B * 5 * 64 * 512;
    nchw2nhwc_k<<<dim3(idivup(tot, 256)), dim3(256), 0, stream>>>(X, XH, B, 5, 64, 512);
  }
  vit_fwd(vit0, XH, 64, 512, 5, 16, VS0);

  fl_fwd(fb1[0], XH, 64, 512, 5, 16, T1, T0, T2);
  fl_fwd(fb1[1], T1, 64, 512, 16, 32, T3, T0, T2);
  fl_fwd(fb1[2], T3, 64, 512, 32, 32, T1, T0, T2);
  {
    long tot = (long)B * 64 * 256 * 32;
    pool12_k<<<dim3(idivup(tot, 256)), dim3(256), 0, stream>>>(T1, DOWN1, B, 64, 512, 32);
  }
  vit_fwd(vit1, DOWN1, 64, 256, 32, 8, VS1);

  fl_fwd(fb2[0], DOWN1, 64, 256, 32, 32, T1, T0, T2);
  fl_fwd(fb2[1], T1, 64, 256, 32, 64, T3, T0, T2);
  fl_fwd(fb2[2], T3, 64, 256, 64, 64, T1, T0, T2);
  {
    long tot = (long)B * 64 * 128 * 64;
    pool12_k<<<dim3(idivup(tot, 256)), dim3(256), 0, stream>>>(T1, DOWN2, B, 64, 256, 64);
  }
  vit_fwd(vit2, DOWN2, 64, 128, 64, 4, VS2);

  dec_fwd(up0, VS0, 16, 8, 4, 8, 512, 512, 256, T0, T1, T3, T2, D0);
  dec_fwd(up1, VS1, 16, 16, 4, 4, 512, 512, 256, T0, T1, T3, T2, D1);
  dec_fwd(up2, VS2, 16, 32, 4, 2, 512, 512, 256, T0, T1, T3, T2, D2);

  // concat [D0|D1|D2] -> T0 (NHWC, 768 ch)
  {
    int rows = B * 64 * 64;
    long tot = (long)rows * 256;
    copy2d_k<<<dim3(idivup(tot, 256), 1, 1), dim3(256), 0, stream>>>(
        T0, 768, 0, D0, 256, 0, rows, 256);
    copy2d_k<<<dim3(idivup(tot, 256), 1, 1), dim3(256), 0, stream>>>(
        T0 + 256, 768, 0, D1, 256, 0, rows, 256);
    copy2d_k<<<dim3(idivup(tot, 256), 1, 1), dim3(256), 0, stream>>>(
        T0 + 512, 768, 0, D2, 256, 0, rows, 256);
  }
  ub_fwd(res1, T0, 64, 64, 768, 256, T1, T3, T2);                    // r3 -> T1
  dec_fwd(fin, T1, 64, 64, 1, 8, 256, 256, 128, T0, T1, T3, T2, T0); // fin -> T0

  // final 1x1 conv to NCLASSES, then NHWC -> NCHW into d_out
  gemm(T0, 128, 0, 0, fc_w, 128, true, 0, 0, FCB, 20, 0, 0,
       nullptr, 0, 0, nullptr, fc_b, B * 64 * 512, 20, 128, 1, 1, 0);
  {
    long tot = (long)B * 20 * 64 * 512;
    nhwc2nchw_k<<<dim3(idivup(tot, 256)), dim3(256), 0, stream>>>(
        FCB, (float*)d_out, B, 20, 64, 512);
  }
}
